// Attention_1400159338855
// MI455X (gfx1250) — matrix operand order repaired and checked by reading, not compiled, not run on hardware
//
#include <hip/hip_runtime.h>
#include <hip/hip_bf16.h>
#include <math.h>

#define P_PIX 4096
#define BATCH 64
#define CDIM  512
#define HDIM  512
#define ADIM  256

typedef __attribute__((ext_vector_type(16))) __bf16 v16bf;
typedef __attribute__((ext_vector_type(8)))  float  v8f;

union BFrag { v16bf v; unsigned int u[8]; };

#if __has_builtin(__builtin_amdgcn_tanhf)
static __device__ __forceinline__ float fast_tanh(float x) {
  return __builtin_amdgcn_tanhf(x);
}
#else
static __device__ __forceinline__ float fast_tanh(float x) { return tanhf(x); }
#endif

// round-to-nearest-even f32 -> bf16, pack two into one dword (lo, hi)
__device__ __forceinline__ unsigned int pack_bf16(float lo, float hi) {
  unsigned int ul = __float_as_uint(lo);
  unsigned int uh = __float_as_uint(hi);
  ul = (ul + 0x7FFFu + ((ul >> 16) & 1u)) >> 16;
  uh = (uh + 0x7FFFu + ((uh >> 16) & 1u)) & 0xFFFF0000u;
  return ul | uh;
}

// ---------------------------------------------------------------------------
// Prep: bias2[b][a] = queries[b]·Ua_w[a] + Ua_b[a] + Wa_b[a];  Wa_w -> bf16
// ---------------------------------------------------------------------------
__global__ void __launch_bounds__(256)
prep_kernel(const float* __restrict__ queries,
            const float* __restrict__ Wa_w,
            const float* __restrict__ Wa_b,
            const float* __restrict__ Ua_w,
            const float* __restrict__ Ua_b,
            float* __restrict__ bias2,
            unsigned int* __restrict__ wa_bf16) {
  const int b = blockIdx.x;      // 64
  const int a = threadIdx.x;     // 256
  const float* q = queries + b * HDIM;
  const float* u = Ua_w + (size_t)a * HDIM;
  float acc = 0.f;
#pragma unroll 8
  for (int h = 0; h < HDIM; ++h) acc += q[h] * u[h];
  bias2[b * ADIM + a] = acc + Wa_b[a] + Ua_b[a];

  // convert Wa_w (256x512 f32) to packed bf16 pairs (65536 dwords)
  const int tid = blockIdx.x * blockDim.x + threadIdx.x;
  for (int i = tid; i < (ADIM * CDIM / 2); i += gridDim.x * blockDim.x)
    wa_bf16[i] = pack_bf16(Wa_w[2 * i], Wa_w[2 * i + 1]);
}

// ---------------------------------------------------------------------------
// Main fused kernel: scores[row] = tanh(keys@Wa^T + bias2)·va + va_b
// rows = p*64+b flattened; 128 rows per block (8 waves x 16 rows), N=256, K=512
// B slabs (256 a x 32 c bf16) staged in LDS, double buffered, 80B row pitch.
// ---------------------------------------------------------------------------
#define BROW_DW 20          // 80 bytes per a-row (64B data + 16B bank pad)
#define BBUF_DW (ADIM * BROW_DW)

__global__ void __launch_bounds__(256)
score_gemm_kernel(const float* __restrict__ keys,
                  const unsigned int* __restrict__ wa_u32,  // packed bf16 pairs
                  const float* __restrict__ bias2,
                  const float* __restrict__ va_w,
                  const float* __restrict__ va_b,
                  float* __restrict__ scores) {
  extern __shared__ float smem[];
  float* sbias = smem;                          // 64*256 f32      (64 KB)
  float* sva   = smem + BATCH * ADIM;           // 256 f32         (1 KB)
  unsigned int* sB0 = (unsigned int*)(smem + BATCH * ADIM + ADIM); // 20 KB
  unsigned int* sB1 = sB0 + BBUF_DW;                               // 20 KB

  const int tid = threadIdx.x;
  // cooperative LDS preload of bias table + va weights
  {
    const float4* gb  = (const float4*)bias2;
    float4*       sb4 = (float4*)sbias;
    for (int i = tid; i < (BATCH * ADIM / 4); i += 256) sb4[i] = gb[i];
    if (tid < ADIM) sva[tid] = va_w[tid];
  }
  // stage first B slab (kk=0): thread tid copies row a=tid (64B)
  {
    const uint4* g = (const uint4*)(wa_u32 + (size_t)tid * (CDIM / 2));
    uint4 r0 = g[0], r1 = g[1], r2 = g[2], r3 = g[3];
    uint4* d = (uint4*)(sB0 + tid * BROW_DW);
    d[0] = r0; d[1] = r1; d[2] = r2; d[3] = r3;
  }
  __syncthreads();

  const int wave = tid >> 5;      // 0..7
  const int lane = tid & 31;
  const int l15  = lane & 15;
  const int lhi  = lane >> 4;     // 0 or 1
  const int rowBase = blockIdx.x * 128 + wave * 16;

  v8f acc[16];
#pragma unroll
  for (int t = 0; t < 16; ++t)
    acc[t] = (v8f){0.f, 0.f, 0.f, 0.f, 0.f, 0.f, 0.f, 0.f};

  // A-frag per-lane source: row = rowBase + (lane&15); K chunks per ISA layout
  const float* arow = keys + (size_t)(rowBase + l15) * CDIM + (lhi * 8);

  for (int kk = 0; kk < CDIM; kk += 32) {
    unsigned int* cur = (kk & 32) ? sB1 : sB0;
    unsigned int* nxt = (kk & 32) ? sB0 : sB1;
    const bool hasNext = (kk + 32) < CDIM;

    // issue next B-slab global fetch early (consumed after the compute block)
    uint4 r0, r1, r2, r3;
    if (hasNext) {
      const uint4* g =
          (const uint4*)(wa_u32 + (size_t)tid * (CDIM / 2) + ((kk + 32) >> 1));
      r0 = g[0]; r1 = g[1]; r2 = g[2]; r3 = g[3];
    }

    // A fragment: 16 bf16 per lane from two 8-float chunks
    BFrag A;
    float4 a0 = *(const float4*)(arow + kk);
    float4 a1 = *(const float4*)(arow + kk + 4);
    float4 a2 = *(const float4*)(arow + kk + 16);
    float4 a3 = *(const float4*)(arow + kk + 20);
    A.u[0] = pack_bf16(a0.x, a0.y); A.u[1] = pack_bf16(a0.z, a0.w);
    A.u[2] = pack_bf16(a1.x, a1.y); A.u[3] = pack_bf16(a1.z, a1.w);
    A.u[4] = pack_bf16(a2.x, a2.y); A.u[5] = pack_bf16(a2.z, a2.w);
    A.u[6] = pack_bf16(a3.x, a3.y); A.u[7] = pack_bf16(a3.z, a3.w);
    if (hasNext) __builtin_prefetch(arow + kk + 32, 0, 0);

#pragma unroll
    for (int t = 0; t < 16; ++t) {
      // B-frag: lane = column a, K chunks 8*lhi..+7 and 16+8*lhi..+7 from LDS slab
      const unsigned int* bq = cur + (t * 16 + l15) * BROW_DW + lhi * 4;
      uint4 b0 = *(const uint4*)bq;
      uint4 b1 = *(const uint4*)(bq + 8);
      BFrag Bf;
      Bf.u[0] = b0.x; Bf.u[1] = b0.y; Bf.u[2] = b0.z; Bf.u[3] = b0.w;
      Bf.u[4] = b1.x; Bf.u[5] = b1.y; Bf.u[6] = b1.z; Bf.u[7] = b1.w;
      acc[t] = __builtin_amdgcn_wmma_f32_16x16x32_bf16(
          false, A.v, false, Bf.v, (short)0, acc[t], false, false);
    }

    if (hasNext) {
      uint4* d = (uint4*)(nxt + tid * BROW_DW);
      d[0] = r0; d[1] = r1; d[2] = r2; d[3] = r3;
    }
    __syncthreads();
  }

  // Epilogue: x -> tanh(x + bias2[b][a]) * va[a], reduce over a (lanes+tiles)
  // C/D layout: element (M = i + 8*lhi, N = l15) in acc[t][i]
  const int wbase = (wave & 3) * 16;      // rows of a block cover b=0..63 twice
  const int bb = wbase + lhi * 8;         // b of row partial i
  const float vb = *va_b;
  float part[8] = {0.f, 0.f, 0.f, 0.f, 0.f, 0.f, 0.f, 0.f};
#pragma unroll
  for (int t = 0; t < 16; ++t) {
    const int a = t * 16 + l15;
    const float va = sva[a];
#pragma unroll
    for (int i = 0; i < 8; ++i) {
      float x = acc[t][i] + sbias[(bb + i) * ADIM + a];
      part[i] += fast_tanh(x) * va;
    }
  }
  // reduce across the 16 lanes that share each row (xor within 16-lane halves)
#pragma unroll
  for (int off = 1; off < 16; off <<= 1) {
#pragma unroll
    for (int i = 0; i < 8; ++i) part[i] += __shfl_xor(part[i], off, 32);
  }
  if (l15 == 0) {
#pragma unroll
    for (int i = 0; i < 8; ++i)
      scores[rowBase + lhi * 8 + i] = part[i] + vb;
  }
}

// ---------------------------------------------------------------------------
// Softmax over p per batch; writes weights to d_out, zeroes context region
// ---------------------------------------------------------------------------
__global__ void __launch_bounds__(256)
softmax_kernel(const float* __restrict__ scores, float* __restrict__ out) {
  const int b = blockIdx.x;
  const int tid = threadIdx.x;
  __shared__ float red[256];

  float lmax = -3.4e38f;
  for (int p = tid; p < P_PIX; p += 256)
    lmax = fmaxf(lmax, scores[p * BATCH + b]);
  red[tid] = lmax; __syncthreads();
  for (int s = 128; s > 0; s >>= 1) {
    if (tid < s) red[tid] = fmaxf(red[tid], red[tid + s]);
    __syncthreads();
  }
  const float m = red[0]; __syncthreads();

  float lsum = 0.f;
  for (int p = tid; p < P_PIX; p += 256)
    lsum += __expf(scores[p * BATCH + b] - m);
  red[tid] = lsum; __syncthreads();
  for (int s = 128; s > 0; s >>= 1) {
    if (tid < s) red[tid] += red[tid + s];
    __syncthreads();
  }
  const float inv = 1.f / red[0];

  float* wout = out + BATCH * CDIM;   // weights section of d_out
  for (int p = tid; p < P_PIX; p += 256)
    wout[p * BATCH + b] = __expf(scores[p * BATCH + b] - m) * inv;

  // zero the context region for this batch (atomically accumulated next)
  out[b * CDIM + tid] = 0.f;
  out[b * CDIM + tid + 256] = 0.f;
}

// ---------------------------------------------------------------------------
// context[b][c] = sum_p w[p][b] * keys[p][b][c]; p split into 8 chunks/batch
// ---------------------------------------------------------------------------
__global__ void __launch_bounds__(256)
context_kernel(const float* __restrict__ keys, float* __restrict__ out) {
  const int chunk = blockIdx.x;   // 0..7  (512 pixels each)
  const int b     = blockIdx.y;   // 0..63
  const int tid   = threadIdx.x;
  const int p0    = chunk * 512;

  __shared__ float sw[512];
  const float* wsrc = out + BATCH * CDIM;
  sw[tid]       = wsrc[(p0 + tid) * BATCH + b];
  sw[tid + 256] = wsrc[(p0 + tid + 256) * BATCH + b];
  __syncthreads();

  float acc0 = 0.f, acc1 = 0.f;
  const float* kb = keys + ((size_t)p0 * BATCH + b) * CDIM;
#pragma unroll 4
  for (int pp = 0; pp < 512; ++pp) {
    const float* kr = kb + (size_t)pp * BATCH * CDIM;
    const float w = sw[pp];
    acc0 += w * kr[tid];
    acc1 += w * kr[tid + 256];
  }
  unsafeAtomicAdd(&out[b * CDIM + tid], acc0);
  unsafeAtomicAdd(&out[b * CDIM + tid + 256], acc1);
}

// ---------------------------------------------------------------------------
extern "C" void kernel_launch(void* const* d_in, const int* in_sizes, int n_in,
                              void* d_out, int out_size, void* d_ws,
                              size_t ws_size, hipStream_t stream) {
  const float* keys    = (const float*)d_in[0];
  const float* queries = (const float*)d_in[1];
  const float* Wa_w    = (const float*)d_in[2];
  const float* Wa_b    = (const float*)d_in[3];
  const float* Ua_w    = (const float*)d_in[4];
  const float* Ua_b    = (const float*)d_in[5];
  const float* va_w    = (const float*)d_in[6];
  const float* va_b    = (const float*)d_in[7];
  float* out = (float*)d_out;

  char* ws = (char*)d_ws;
  float*        bias2   = (float*)ws;                       // 64*256*4   = 64 KB
  unsigned int* wa_bf16 = (unsigned int*)(ws + 65536);      // 256*512*2  = 256 KB
  float*        scores  = (float*)(ws + 65536 + 262144);    // 262144*4   = 1 MB

  prep_kernel<<<64, 256, 0, stream>>>(queries, Wa_w, Wa_b, Ua_w, Ua_b,
                                      bias2, wa_bf16);

  const size_t smem1 =
      (size_t)(BATCH * ADIM + ADIM) * sizeof(float) +      // bias + va
      (size_t)2 * BBUF_DW * sizeof(unsigned int);          // B double buffer
  score_gemm_kernel<<<2048, 256, smem1, stream>>>(
      keys, wa_bf16, bias2, va_w, va_b, scores);

  softmax_kernel<<<64, 256, 0, stream>>>(scores, out);

  context_kernel<<<dim3(8, 64), 256, 0, stream>>>(keys, out);
}
